// SAttention_42769284334294
// MI455X (gfx1250) — 8-bit matrix operand order repaired and checked by reading, compiled, not run on hardware
//
#include <hip/hip_runtime.h>
#include <math.h>


typedef float v2f __attribute__((ext_vector_type(2)));
typedef float v8f __attribute__((ext_vector_type(8)));
typedef int   v8i __attribute__((ext_vector_type(8)));

#define B_ 4
#define N_ 1024
#define C_ 512
#define H_ 8
#define D_ 64

template <typename P>
__device__ __forceinline__ v8i load_a64(const P* p) {
    v8i A;
    A[0] = *(const int*)(p + 0);
    A[1] = *(const int*)(p + 4);
    A[2] = *(const int*)(p + 8);
    A[3] = *(const int*)(p + 12);
    A[4] = *(const int*)(p + 32);
    A[5] = *(const int*)(p + 36);
    A[6] = *(const int*)(p + 40);
    A[7] = *(const int*)(p + 44);
    return A;
}
template <typename P>
__device__ __forceinline__ v8i load_b64(const P* p) {
    int4 lo = *(const int4*)(p);
    int4 hi = *(const int4*)(p + 32);
    v8i Bv;
    Bv[0] = lo.x; Bv[1] = lo.y; Bv[2] = lo.z; Bv[3] = lo.w;
    Bv[4] = hi.x; Bv[5] = hi.y; Bv[6] = hi.z; Bv[7] = hi.w;
    return Bv;
}

__device__ __forceinline__ int if_spike(float v, int a0) {
    if (v >= 1.0f && a0 < 7)  return 1;
    if (v <  0.0f && a0 > -8) return -1;
    return 0;
}

__device__ __forceinline__ float hmax16(float v) {
    #pragma unroll
    for (int off = 1; off < 16; off <<= 1) v = fmaxf(v, __shfl_xor(v, off, 32));
    return v;
}
__device__ __forceinline__ float hsum16(float v) {
    #pragma unroll
    for (int off = 1; off < 16; off <<= 1) v += __shfl_xor(v, off, 32);
    return v;
}

__global__ void k1_qkv_if_pack(const float* __restrict__ x,
                               const float* __restrict__ w_qkv,
                               const float* __restrict__ b_qkv,
                               const float* __restrict__ q_vmem,
                               const float* __restrict__ k_vmem,
                               const float* __restrict__ v_vmem,
                               const int* __restrict__ q_acc0,
                               const int* __restrict__ k_acc0,
                               const int* __restrict__ v_acc0,
                               char* __restrict__ qA,
                               char* __restrict__ kB,
                               char* __restrict__ vB0,
                               char* __restrict__ vB1)
{
    const int lane = threadIdx.x & 31;
    const int half = lane >> 4;
    const int lm   = lane & 15;
    const int wave = (blockIdx.x * blockDim.x + threadIdx.x) >> 5;
    const int tileM = wave / 24;
    const int tileC = wave % 24;

    const float* xr[2];
    const float* wr[4];
    #pragma unroll
    for (int i = 0; i < 2; ++i)
        xr[i] = x + (size_t)(tileM * 32 + i * 16 + lm) * C_;
    #pragma unroll
    for (int i = 0; i < 4; ++i)
        wr[i] = w_qkv + (size_t)(tileC * 64 + i * 16 + lm) * C_;

    v8f cf[2][4];
    #pragma unroll
    for (int mi = 0; mi < 2; ++mi)
        #pragma unroll
        for (int ci = 0; ci < 4; ++ci) cf[mi][ci] = (v8f){};

    for (int k = 0; k < C_; k += 4) {
        v2f a[2], b[4];
        #pragma unroll
        for (int i = 0; i < 2; ++i) a[i] = *(const v2f*)(xr[i] + k + 2 * half);
        #pragma unroll
        for (int i = 0; i < 4; ++i) b[i] = *(const v2f*)(wr[i] + k + 2 * half);
        #pragma unroll
        for (int mi = 0; mi < 2; ++mi)
            #pragma unroll
            for (int ci = 0; ci < 4; ++ci)
                cf[mi][ci] = __builtin_amdgcn_wmma_f32_16x16x4_f32(
                    false, a[mi], false, b[ci], (short)0, cf[mi][ci], false, false);
    }

    const int colB64 = tileC * 64;
    const int s = colB64 / 512;
    const int h = (colB64 % 512) / 64;
    const float* vmem = (s == 0) ? q_vmem : (s == 1) ? k_vmem : v_vmem;
    const int*   acc0 = (s == 0) ? q_acc0 : (s == 1) ? k_acc0 : v_acc0;

    #pragma unroll
    for (int ci = 0; ci < 4; ++ci) {
        const int d = ci * 16 + lm;
        const float bias = b_qkv[colB64 + ci * 16 + lm];
        #pragma unroll
        for (int mi = 0; mi < 2; ++mi) {
            #pragma unroll
            for (int g = 0; g < 8; ++g) {
                const int r = tileM * 32 + mi * 16 + g + 8 * half;
                const int b = r >> 10;
                const int n = r & 1023;
                const size_t bhn = ((size_t)b * H_ + h) * N_ + n;
                const size_t i4  = bhn * D_ + d;
                const float val  = cf[mi][ci][g] + bias;
                const int   a0   = acc0[i4];
                const int   sp   = if_spike(vmem[i4] + val, a0);
                if (s == 0) {
                    qA[bhn * 128 + d]      = (char)(a0 + sp);
                    qA[bhn * 128 + 64 + d] = (char)sp;
                } else if (s == 1) {
                    kB[bhn * 128 + d]      = (char)sp;
                    kB[bhn * 128 + 64 + d] = (char)a0;
                } else {
                    const size_t cp = (((size_t)b * H_ + h) * D_ + d) * N_ + n;
                    vB0[cp] = (char)sp;
                    vB1[cp] = (char)a0;
                }
            }
        }
    }
}

__global__ void k2_attn_softmax_if(const char* __restrict__ qA,
                                   const char* __restrict__ kB,
                                   const float* __restrict__ attn_vmem,
                                   const int* __restrict__ attn_acc0,
                                   char* __restrict__ attn_s,
                                   char* __restrict__ attn_acc)
{
    __shared__ float pmx[16][8];
    __shared__ float psm[16][8];

    const int tid  = threadIdx.x;
    const int lane = tid & 31;
    const int half = lane >> 4;
    const int lm   = lane & 15;
    const int w    = tid >> 5;

    const int bh      = blockIdx.x >> 6;
    const int rowBase = (blockIdx.x & 63) * 16;

    const char* arow = qA + ((size_t)bh * N_ + rowBase + lm) * 128;
    const v8i A0 = load_a64(arow + 0  + half * 16);
    const v8i A1 = load_a64(arow + 64 + half * 16);

    float scf[8][8];
    #pragma unroll
    for (int i = 0; i < 8; ++i) {
        const int colBase = (w + 8 * i) * 16;
        const char* brow = kB + ((size_t)bh * N_ + colBase + lm) * 128;
        const v8i B0 = load_b64(brow + 0  + half * 16);
        const v8i B1 = load_b64(brow + 64 + half * 16);
        v8i acc = {};
        acc = __builtin_amdgcn_wmma_i32_16x16x64_iu8(true, A0, true, B0, acc, false, false);
        acc = __builtin_amdgcn_wmma_i32_16x16x64_iu8(true, A1, true, B1, acc, false, false);
        #pragma unroll
        for (int g = 0; g < 8; ++g) scf[i][g] = 0.125f * (float)acc[g];
    }

    float rm[8];
    #pragma unroll
    for (int g = 0; g < 8; ++g) {
        float pm = scf[0][g];
        #pragma unroll
        for (int i = 1; i < 8; ++i) pm = fmaxf(pm, scf[i][g]);
        pm = hmax16(pm);
        if (lm == 0) pmx[g + 8 * half][w] = pm;
    }
    __syncthreads();
    #pragma unroll
    for (int g = 0; g < 8; ++g) {
        float m = pmx[g + 8 * half][0];
        #pragma unroll
        for (int j = 1; j < 8; ++j) m = fmaxf(m, pmx[g + 8 * half][j]);
        rm[g] = m;
    }

    float inv[8];
    #pragma unroll
    for (int g = 0; g < 8; ++g) {
        float ps = 0.f;
        #pragma unroll
        for (int i = 0; i < 8; ++i) {
            const float e = expf(scf[i][g] - rm[g]);
            scf[i][g] = e;
            ps += e;
        }
        ps = hsum16(ps);
        if (lm == 0) psm[g + 8 * half][w] = ps;
    }
    __syncthreads();
    #pragma unroll
    for (int g = 0; g < 8; ++g) {
        float s = 0.f;
        #pragma unroll
        for (int j = 0; j < 8; ++j) s += psm[g + 8 * half][j];
        inv[g] = 7.0f / s;
    }

    #pragma unroll
    for (int g = 0; g < 8; ++g) {
        const int grow = rowBase + g + 8 * half;
        const size_t rb = ((size_t)bh * N_ + grow) * (size_t)N_;
        #pragma unroll
        for (int i = 0; i < 8; ++i) {
            const int col = (w + 8 * i) * 16 + lm;
            const float p  = scf[i][g] * inv[g];
            const int   a0 = attn_acc0[rb + col];
            const int   sp = if_spike(attn_vmem[rb + col] + 0.2f + p, a0);
            attn_s[rb + col]   = (char)sp;
            attn_acc[rb + col] = (char)(a0 + sp);
        }
    }
}

__global__ void k3_out_gemm_if(const char* __restrict__ attn_acc,
                               const char* __restrict__ attn_s,
                               const char* __restrict__ vB0,
                               const char* __restrict__ vB1,
                               float* __restrict__ out2)
{
    extern __shared__ char lds[];
    char* vbuf0 = lds;
    char* vbuf1 = lds + 65536;

    const int tid  = threadIdx.x;
    const int lane = tid & 31;
    const int half = lane >> 4;
    const int lm   = lane & 15;
    const int w    = tid >> 5;

    const int bh = blockIdx.x >> 3;
    const int sg = blockIdx.x & 7;

    {
        const int4* g0 = (const int4*)(vB0 + (size_t)bh * 65536);
        const int4* g1 = (const int4*)(vB1 + (size_t)bh * 65536);
        int4* s0 = (int4*)vbuf0;
        int4* s1 = (int4*)vbuf1;
        for (int i = tid; i < 4096; i += 256) {
            s0[i] = g0[i];
            s1[i] = g1[i];
        }
    }
    __syncthreads();

    const int mt = sg * 8 + w;
    const char* aaRow = attn_acc + ((size_t)bh * N_ + mt * 16 + lm) * N_;
    const char* asRow = attn_s   + ((size_t)bh * N_ + mt * 16 + lm) * N_;

    v8i c[4] = {{}, {}, {}, {}};
    for (int kc = 0; kc < N_; kc += 64) {
        const v8i Aa = load_a64(aaRow + kc + half * 16);
        const v8i As = load_a64(asRow + kc + half * 16);
        #pragma unroll
        for (int nt = 0; nt < 4; ++nt) {
            const v8i Bs = load_b64(vbuf0 + (nt * 16 + lm) * 1024 + kc + half * 16);
            const v8i Bc = load_b64(vbuf1 + (nt * 16 + lm) * 1024 + kc + half * 16);
            c[nt] = __builtin_amdgcn_wmma_i32_16x16x64_iu8(true, Aa, true, Bs, c[nt], false, false);
            c[nt] = __builtin_amdgcn_wmma_i32_16x16x64_iu8(true, As, true, Bc, c[nt], false, false);
        }
    }

    const int b = bh >> 3, h = bh & 7;
    #pragma unroll
    for (int nt = 0; nt < 4; ++nt) {
        #pragma unroll
        for (int g = 0; g < 8; ++g) {
            const int n = mt * 16 + g + 8 * half;
            const float val = (float)c[nt][g];
            const float o = (val >= 0.5f ? 1.0f : 0.0f) - (val < -0.5f ? 1.0f : 0.0f);
            out2[((size_t)b * N_ + n) * C_ + h * D_ + nt * 16 + lm] = o;
        }
    }
}

__global__ void k4_proj(const float* __restrict__ out2,
                        const float* __restrict__ w_proj,
                        const float* __restrict__ b_proj,
                        float* __restrict__ out)
{
    const int lane = threadIdx.x & 31;
    const int half = lane >> 4;
    const int lm   = lane & 15;
    const int wave = (blockIdx.x * blockDim.x + threadIdx.x) >> 5;
    const int tileM = wave >> 3;
    const int tileC = wave & 7;

    const float* xr[2];
    const float* wr[4];
    #pragma unroll
    for (int i = 0; i < 2; ++i)
        xr[i] = out2 + (size_t)(tileM * 32 + i * 16 + lm) * C_;
    #pragma unroll
    for (int i = 0; i < 4; ++i)
        wr[i] = w_proj + (size_t)(tileC * 64 + i * 16 + lm) * C_;

    v8f cf[2][4];
    #pragma unroll
    for (int mi = 0; mi < 2; ++mi)
        #pragma unroll
        for (int ci = 0; ci < 4; ++ci) cf[mi][ci] = (v8f){};

    for (int k = 0; k < C_; k += 4) {
        v2f a[2], b[4];
        #pragma unroll
        for (int i = 0; i < 2; ++i) a[i] = *(const v2f*)(xr[i] + k + 2 * half);
        #pragma unroll
        for (int i = 0; i < 4; ++i) b[i] = *(const v2f*)(wr[i] + k + 2 * half);
        #pragma unroll
        for (int mi = 0; mi < 2; ++mi)
            #pragma unroll
            for (int ci = 0; ci < 4; ++ci)
                cf[mi][ci] = __builtin_amdgcn_wmma_f32_16x16x4_f32(
                    false, a[mi], false, b[ci], (short)0, cf[mi][ci], false, false);
    }

    #pragma unroll
    for (int ci = 0; ci < 4; ++ci) {
        const int col = tileC * 64 + ci * 16 + lm;
        const float bias = b_proj[col];
        #pragma unroll
        for (int mi = 0; mi < 2; ++mi) {
            #pragma unroll
            for (int g = 0; g < 8; ++g) {
                const int r = tileM * 32 + mi * 16 + g + 8 * half;
                out[(size_t)r * C_ + col] = cf[mi][ci][g] + bias;
            }
        }
    }
}

extern "C" void kernel_launch(void* const* d_in, const int* in_sizes, int n_in,
                              void* d_out, int out_size, void* d_ws, size_t ws_size,
                              hipStream_t stream)
{
    const float* x         = (const float*)d_in[0];
    const float* w_qkv     = (const float*)d_in[1];
    const float* b_qkv     = (const float*)d_in[2];
    const float* w_proj    = (const float*)d_in[3];
    const float* b_proj    = (const float*)d_in[4];
    const float* q_vmem    = (const float*)d_in[5];
    const float* k_vmem    = (const float*)d_in[6];
    const float* v_vmem    = (const float*)d_in[7];
    const float* attn_vmem = (const float*)d_in[8];
    const int*   q_acc0    = (const int*)d_in[9];
    const int*   k_acc0    = (const int*)d_in[10];
    const int*   v_acc0    = (const int*)d_in[11];
    const int*   attn_acc0 = (const int*)d_in[12];

    char* ws   = (char*)d_ws;
    char* qA   = ws;
    char* kB   = qA  + (size_t)4  * 1024 * 1024;
    char* vB0  = kB  + (size_t)4  * 1024 * 1024;
    char* vB1  = vB0 + (size_t)2  * 1024 * 1024;
    char* at_s = vB1 + (size_t)2  * 1024 * 1024;
    char* at_a = at_s + (size_t)32 * 1024 * 1024;
    float* out2 = (float*)(at_a + (size_t)32 * 1024 * 1024);

    k1_qkv_if_pack<<<384, 256, 0, stream>>>(x, w_qkv, b_qkv,
                                            q_vmem, k_vmem, v_vmem,
                                            q_acc0, k_acc0, v_acc0,
                                            qA, kB, vB0, vB1);
    k2_attn_softmax_if<<<2048, 256, 0, stream>>>(qA, kB, attn_vmem, attn_acc0,
                                                 at_s, at_a);
    k3_out_gemm_if<<<256, 256, 131072, stream>>>(at_a, at_s, vB0, vB1, out2);
    k4_proj<<<128, 256, 0, stream>>>(out2, w_proj, b_proj, (float*)d_out);
}
